// PTBModel_55147380081279
// MI455X (gfx1250) — matrix operand order repaired and checked by reading, not compiled, not run on hardware
//
#include <hip/hip_runtime.h>
#include <hip/hip_bf16.h>

// ---------------------------------------------------------------------------
// PTB LSTM LM on MI455X (gfx1250, wave32):
//   embed -> [x@W0+b0 big GEMM] -> seq scan (h@U0 WMMA + fused gates)
//         -> [h@W1+b1 big GEMM] -> seq scan (h@U1)
//         -> [h@W_out + b_out GEMM, row-remapped]
// All GEMMs use v_wmma_f32_16x16x32_f16 (f16 in, f32 accum).
// WMMA kernels are branch-free (exact grids, template strips) so EXEC stays
// all-1s. Inner loops: rolled 2-stage ping-pong (two live fragment sets,
// loop-carried set reloaded in place) so load latency overlaps WMMA issue
// without register copies; scan kernel relaxes occupancy so RA keeps both
// sets resident.
// ---------------------------------------------------------------------------

typedef __attribute__((ext_vector_type(16))) _Float16 v16h;
typedef __attribute__((ext_vector_type(8)))  _Float16 v8h;
typedef __attribute__((ext_vector_type(8)))  float    v8f;

__device__ __forceinline__ v8f wmma_f16(v16h a, v16h b, v8f c) {
  // 8 args: (neg_a, A, neg_b, B, c_mod, C, reuse_a, reuse_b)
  return __builtin_amdgcn_wmma_f32_16x16x32_f16(false, a, false, b, (short)0, c,
                                                false, false);
}

// A-fragment (16x32 f16, row-major A[M][K]): lane-half h holds row (lane&15),
// contiguous K = k0+16h .. k0+16h+15 (matches B-fragment order).
__device__ __forceinline__ v16h load_a_frag(const _Float16* __restrict__ Arow,
                                            int k0, int halfsel) {
  v8h lo = *(const v8h*)(Arow + k0 + 16 * halfsel);
  v8h hi = *(const v8h*)(Arow + k0 + 16 * halfsel + 8);
  return __builtin_shufflevector(lo, hi, 0, 1, 2, 3, 4, 5, 6, 7,
                                 8, 9, 10, 11, 12, 13, 14, 15);
}

// B-fragment (32x16 f16) from pre-transposed Bt[N][K]: lane-half h holds
// column (lane&15), contiguous K = k0+16h .. k0+16h+15.
__device__ __forceinline__ v16h load_b_frag(const _Float16* __restrict__ Btrow,
                                            int k0, int halfsel) {
  return *(const v16h*)(Btrow + k0 + 16 * halfsel);
}

// One k-step worth of operands: A fragment + S B fragments.
template <int S>
struct Frags {
  v16h a;
  v16h b[S];
};

template <int S>
__device__ __forceinline__ Frags<S> load_frags(
    const _Float16* __restrict__ Arow,
    const _Float16* __restrict__ const* Brow, int k0, int halfsel) {
  Frags<S> f;
  f.a = load_a_frag(Arow, k0, halfsel);
#pragma unroll
  for (int s = 0; s < S; ++s) f.b[s] = load_b_frag(Brow[s], k0, halfsel);
  return f;
}

template <int S>
__device__ __forceinline__ void mma_frags(v8f (&acc)[S], const Frags<S>& f) {
#pragma unroll
  for (int s = 0; s < S; ++s) acc[s] = wmma_f16(f.a, f.b[s], acc[s]);
}

// Ping-pong K loop: requires K % 64 == 0, K >= 64. f0/f1 are distinct named
// sets; f0 is loop-carried and RELOADED in place (loads write its PHI regs,
// no copies). Kept rolled (#pragma unroll 1) so the two sets stay live and
// the scheduler overlaps one set's loads with the other set's WMMAs.
template <int S>
__device__ __forceinline__ void k_loop_pingpong(
    v8f (&acc)[S], const _Float16* __restrict__ Arow,
    const _Float16* __restrict__ const* Brow, int K, int halfsel) {
  Frags<S> f0 = load_frags<S>(Arow, Brow, 0, halfsel);
#pragma unroll 1
  for (int k0 = 0; k0 < K - 64; k0 += 64) {
    Frags<S> f1 = load_frags<S>(Arow, Brow, k0 + 32, halfsel);
    mma_frags<S>(acc, f0);
    f0 = load_frags<S>(Arow, Brow, k0 + 64, halfsel);  // fresh loads into f0
    mma_frags<S>(acc, f1);
  }
  Frags<S> f1 = load_frags<S>(Arow, Brow, K - 32, halfsel);
  mma_frags<S>(acc, f0);
  mma_frags<S>(acc, f1);
}

// ---------------------------------------------------------------------------
// Weight convert + transpose: src fp32 [K][N] -> dst f16 [N][K]
// ---------------------------------------------------------------------------
__global__ void convert_transpose(const float* __restrict__ src,
                                  _Float16* __restrict__ dst, int K, int N) {
  int idx = blockIdx.x * blockDim.x + threadIdx.x;
  if (idx >= K * N) return;
  int n = idx % N;          // coalesced read along N
  int k = idx / N;
  dst[(size_t)n * K + k] = (_Float16)src[idx];
}

// Embedding gather: x0[t*B + b][E] (f16), tokens[b][t]
__global__ void embed_gather(const int* __restrict__ tokens,
                             const float* __restrict__ emb,
                             _Float16* __restrict__ x0, int Bsz, int T, int E) {
  int row = blockIdx.x;                 // t*B + b
  int t = row / Bsz, b = row % Bsz;
  int tok = tokens[b * T + t];
  const float* s = emb + (size_t)tok * E;
  _Float16* d = x0 + (size_t)row * E;
  for (int e = threadIdx.x; e < E; e += blockDim.x) d[e] = (_Float16)s[e];
}

__global__ void zero_f16(_Float16* __restrict__ p, int n) {
  int i = blockIdx.x * blockDim.x + threadIdx.x;
  if (i < n) p[i] = (_Float16)0.0f;
}
__global__ void zero_f32(float* __restrict__ p, int n) {
  int i = blockIdx.x * blockDim.x + threadIdx.x;
  if (i < n) p[i] = 0.0f;
}

// ---------------------------------------------------------------------------
// WMMA GEMM: C[M][N] f32 = A[M][K] f16 @ Bt[N][K]^T f16 + bias[N].
// One wave computes a 16 x (16*S) strip (S accumulators, A-fragment reuse xS).
// REQUIREMENTS (enforced by the launcher): N % (16*S) == 0, M % 16 == 0,
// grid covers exactly (M/16)*(N/(16*S)) waves; DB additionally needs K%64==0.
// DB: ping-pong double buffering (use when occupancy is low).
// REMAP: store row (t*32+b) -> (b*T + t) to match reference row order.
// ---------------------------------------------------------------------------
template <int S, bool REMAP, bool DB>
__global__ __launch_bounds__(256) void gemm_f16_strip(
    const _Float16* __restrict__ A, const _Float16* __restrict__ Bt,
    const float* __restrict__ bias, float* __restrict__ C,
    int M, int N, int K, int remapT) {
  const int lane = threadIdx.x & 31;
  const int wid  = blockIdx.x * (blockDim.x >> 5) + (threadIdx.x >> 5);
  const int NS = N / (16 * S);
  const int ms = wid / NS;
  const int ns = wid % NS;
  const int m0 = ms << 4;
  const int halfsel = lane >> 4;
  const int lm = lane & 15;

  v8f acc[S];
#pragma unroll
  for (int s = 0; s < S; ++s) acc[s] = (v8f){0, 0, 0, 0, 0, 0, 0, 0};

  const _Float16* Arow = A + (size_t)(m0 + lm) * K;
  const _Float16* Brow[S];
#pragma unroll
  for (int s = 0; s < S; ++s)
    Brow[s] = Bt + (size_t)((ns * S + s) * 16 + lm) * K;

  if (DB) {
    k_loop_pingpong<S>(acc, Arow, Brow, K, halfsel);
  } else {
    // Grouped loads: one clause + one wait per k-step, S back-to-back WMMAs.
    for (int k0 = 0; k0 < K; k0 += 32) {
      Frags<S> f = load_frags<S>(Arow, Brow, k0, halfsel);
      mma_frags<S>(acc, f);
    }
  }

#pragma unroll
  for (int s = 0; s < S; ++s) {
    int n = (ns * S + s) * 16 + lm;
    float bv = bias[n];
#pragma unroll
    for (int r = 0; r < 8; ++r) {
      int m = m0 + r + (halfsel << 3);
      int row = m;
      if (REMAP) { int t = m >> 5; int b = m & 31; row = b * remapT + t; }
      C[(size_t)row * N + n] = acc[s][r] + bv;
    }
  }
}

// ---------------------------------------------------------------------------
// One LSTM timestep, fully fused:
//   z = h_prev[32][512] @ U^T (Ut[2048][512]) + xw_t[32][2048]  (bias folded)
//   i,f,g,o gates -> c,h update; h written as f16 for next step / next layer.
// Exactly 64 waves (8 blocks x 8 waves): wave owns the same 16x16 (m,j) tile
// of all four gates. ~2 waves/SIMD alive by construction, so relax the
// occupancy target (waves_per_eu 1) letting RA keep both ping-pong fragment
// sets resident instead of folding them onto one register buffer.
// ---------------------------------------------------------------------------
__global__ __launch_bounds__(256)
__attribute__((amdgpu_waves_per_eu(1, 4)))
void lstm_step(
    const _Float16* __restrict__ h_prev,  // [32][512]
    const _Float16* __restrict__ Ut,      // [2048][512]
    const float* __restrict__ xw_t,       // [32][2048]
    float* __restrict__ c_state,          // [32][512]
    _Float16* __restrict__ h_out) {       // [32][512]
  const int lane = threadIdx.x & 31;
  const int wid  = blockIdx.x * (blockDim.x >> 5) + (threadIdx.x >> 5);
  const int mt = wid >> 5;        // 0..1  (rows 0-15 / 16-31)
  const int jt = wid & 31;        // 0..31 (H columns, 16 each)
  const int m0 = mt << 4;
  const int halfsel = lane >> 4;
  const int lm = lane & 15;
  const int j = (jt << 4) + lm;   // column within H owned by this lane

  // Seed accumulators with xw (= x@W + b) for the four gates at (m, j).
  v8f acc[4];
#pragma unroll
  for (int g = 0; g < 4; ++g) {
#pragma unroll
    for (int r = 0; r < 8; ++r) {
      int m = m0 + r + (halfsel << 3);
      acc[g][r] = xw_t[(size_t)m * 2048 + g * 512 + j];
    }
  }

  const _Float16* Arow = h_prev + (size_t)(m0 + lm) * 512;
  const _Float16* Brow[4];
#pragma unroll
  for (int g = 0; g < 4; ++g)
    Brow[g] = Ut + (size_t)(g * 512 + (jt << 4) + lm) * 512;

  k_loop_pingpong<4>(acc, Arow, Brow, 512, halfsel);

#pragma unroll
  for (int r = 0; r < 8; ++r) {
    int m = m0 + r + (halfsel << 3);
    size_t idx = (size_t)m * 512 + j;
    float iv = 1.0f / (1.0f + __expf(-acc[0][r]));
    float fv = 1.0f / (1.0f + __expf(-acc[1][r]));
    float gv = tanhf(acc[2][r]);
    float ov = 1.0f / (1.0f + __expf(-acc[3][r]));
    float c_new = fv * c_state[idx] + iv * gv;
    c_state[idx] = c_new;
    h_out[idx] = (_Float16)(ov * tanhf(c_new));
  }
}

// ---------------------------------------------------------------------------
extern "C" void kernel_launch(void* const* d_in, const int* in_sizes, int n_in,
                              void* d_out, int out_size, void* d_ws,
                              size_t ws_size, hipStream_t stream) {
  (void)in_sizes; (void)n_in; (void)out_size; (void)ws_size;
  const int*   tokens = (const int*)d_in[0];
  const float* emb    = (const float*)d_in[1];
  const float* W0     = (const float*)d_in[2];
  const float* U0     = (const float*)d_in[3];
  const float* b0     = (const float*)d_in[4];
  const float* W1     = (const float*)d_in[5];
  const float* U1     = (const float*)d_in[6];
  const float* b1     = (const float*)d_in[7];
  const float* Wo     = (const float*)d_in[8];
  const float* bo     = (const float*)d_in[9];
  float* out = (float*)d_out;

  constexpr int Bz = 32, T = 256, E = 512, H = 512, V = 10000;
  constexpr int M = Bz * T;           // 8192 rows in [t*B + b] order
  constexpr int G4 = 4 * H;           // 2048

  char* ws = (char*)d_ws;
  size_t off = 0;
  auto take = [&](size_t bytes) -> char* {
    char* p = ws + off;
    off += (bytes + 255) & ~(size_t)255;
    return p;
  };
  _Float16* Wt0 = (_Float16*)take((size_t)G4 * E * 2);      // [2048][512]
  _Float16* Ut0 = (_Float16*)take((size_t)G4 * H * 2);
  _Float16* Wt1 = (_Float16*)take((size_t)G4 * H * 2);
  _Float16* Ut1 = (_Float16*)take((size_t)G4 * H * 2);
  _Float16* Wot = (_Float16*)take((size_t)V * H * 2);       // [10000][512]
  _Float16* x0  = (_Float16*)take((size_t)M * E * 2);       // [t*B+b][E]
  float*    xw  = (float*)take((size_t)M * G4 * 4);         // reused per layer
  _Float16* h0  = (_Float16*)take((size_t)(T + 1) * Bz * H * 2);
  _Float16* h1  = (_Float16*)take((size_t)(T + 1) * Bz * H * 2);
  float*    cst = (float*)take((size_t)Bz * H * 4);

  const int thr = 256;
  // Weight conversion (fp32 -> f16, transposed so B-fragments load contiguous)
  convert_transpose<<<(E * G4 + thr - 1) / thr, thr, 0, stream>>>(W0, Wt0, E, G4);
  convert_transpose<<<(H * G4 + thr - 1) / thr, thr, 0, stream>>>(U0, Ut0, H, G4);
  convert_transpose<<<(H * G4 + thr - 1) / thr, thr, 0, stream>>>(W1, Wt1, H, G4);
  convert_transpose<<<(H * G4 + thr - 1) / thr, thr, 0, stream>>>(U1, Ut1, H, G4);
  convert_transpose<<<(H * V  + thr - 1) / thr, thr, 0, stream>>>(Wo, Wot, H, V);

  embed_gather<<<M, 128, 0, stream>>>(tokens, emb, x0, Bz, T, E);

  // Hidden GEMMs: N=2048, strip S=4 (64 cols) -> NS=32, waves=512*32=16384
  //   -> 2048 blocks * 8 waves, EXACT (no guard needed).
  // Output GEMM:  N=10000, strip S=5 (80 cols) -> NS=125, waves=512*125=64000
  //   -> 8000 blocks * 8 waves, EXACT.
  constexpr int GEMM_H_BLOCKS   = (M / 16) * (G4 / 64) / 8;  // 2048
  constexpr int GEMM_OUT_BLOCKS = (M / 16) * (V / 80) / 8;   // 8000

  // ---- layer 0 ----
  zero_f16<<<(Bz * H + thr - 1) / thr, thr, 0, stream>>>(h0, Bz * H);  // h_{-1}=0
  zero_f32<<<(Bz * H + thr - 1) / thr, thr, 0, stream>>>(cst, Bz * H); // c_{-1}=0
  gemm_f16_strip<4, false, true><<<GEMM_H_BLOCKS, thr, 0, stream>>>(
      x0, Wt0, b0, xw, M, G4, E, 0);                                   // xw = x@W0+b0
  for (int t = 0; t < T; ++t)
    lstm_step<<<8, 256, 0, stream>>>(h0 + (size_t)t * Bz * H, Ut0,
                                     xw + (size_t)t * Bz * G4, cst,
                                     h0 + (size_t)(t + 1) * Bz * H);

  // ---- layer 1 ----
  zero_f16<<<(Bz * H + thr - 1) / thr, thr, 0, stream>>>(h1, Bz * H);
  zero_f32<<<(Bz * H + thr - 1) / thr, thr, 0, stream>>>(cst, Bz * H);
  gemm_f16_strip<4, false, true><<<GEMM_H_BLOCKS, thr, 0, stream>>>(
      h0 + (size_t)Bz * H, Wt1, b1, xw, M, G4, H, 0);                  // h0@W1+b1
  for (int t = 0; t < T; ++t)
    lstm_step<<<8, 256, 0, stream>>>(h1 + (size_t)t * Bz * H, Ut1,
                                     xw + (size_t)t * Bz * G4, cst,
                                     h1 + (size_t)(t + 1) * Bz * H);

  // ---- output projection, remap rows (t*B+b) -> (b*T+t) ----
  // High occupancy hides latency; keep VGPRs low (no double buffer).
  gemm_f16_strip<5, true, false><<<GEMM_OUT_BLOCKS, thr, 0, stream>>>(
      h1 + (size_t)Bz * H, Wot, bo, out, M, V, H, T);
}
